// MLA_82635170775588
// MI455X (gfx1250) — compile-verified, matrix operand order checked by reading, not run on hardware
//
#include <hip/hip_runtime.h>
#include <hip/hip_bf16.h>

// ---------------------------------------------------------------------------
// MLA forward for MI455X (gfx1250, wave32, WMMA + Tensor Data Mover).
// - All GEMMs + attention matmuls: v_wmma_f32_16x16x32_bf16 (fp32 accum).
// - Big GEMMs: TDM (tensor_load_to_lds) double-buffered LDS staging,
//   TENSORcnt-synchronized, with TDM padding for bank-conflict-free reads.
// - Attention: streaming online-softmax (never materializes S x S).
// Workspace: ~216 MB.
// ---------------------------------------------------------------------------

#define Hn    16
#define NOPE  128
#define ROPEd 64
#define VDd   128
#define QKD   192
#define QRr   1536
#define KRr   512
#define Bsz   2
#define Ssz   2048
#define Dm    2048
#define Mrows (Bsz * Ssz)          // 4096

typedef __bf16 bf16;
typedef __attribute__((ext_vector_type(16))) __bf16 v16bf;
typedef __attribute__((ext_vector_type(8)))  __bf16 v8bf;
typedef __attribute__((ext_vector_type(8)))  float  v8f;
typedef __attribute__((ext_vector_type(4))) unsigned int u32x4;
typedef __attribute__((ext_vector_type(8))) int         i32x8;
typedef __attribute__((ext_vector_type(4))) int         i32x4;

__device__ __forceinline__ bf16 f2bf(float f) {
  union { float f; unsigned int u; } a; a.f = f;
  unsigned int r = a.u + 0x7FFFu + ((a.u >> 16) & 1u);   // RNE
  union { unsigned short u; bf16 b; } o; o.u = (unsigned short)(r >> 16);
  return o.b;
}

__device__ __forceinline__ v8f v8f_zero() {
  v8f z; for (int i = 0; i < 8; ++i) z[i] = 0.f; return z;
}

__device__ __forceinline__ v8f wmma_bf16(v16bf a, v16bf b, v8f c) {
  return __builtin_amdgcn_wmma_f32_16x16x32_bf16(
      false, a, false, b, (short)0, c, false, false);
}

// 16x32 bf16 fragment in ISA A/B layout from a row-major tile (stride ldk).
// lanes 0-15: row=lane, K{0..7}+{16..23}; lanes 16-31: row=lane-16, K{8..15}+{24..31}
__device__ __forceinline__ v16bf load_frag_nt(const bf16* __restrict__ base,
                                              int ldk, int lane) {
  int row = lane & 15;
  int ko  = (lane >> 4) * 8;
  const bf16* p = base + (size_t)row * ldk + ko;
  union { v16bf v; v8bf h[2]; } u;
  u.h[0] = *(const v8bf*)(p);
  u.h[1] = *(const v8bf*)(p + 16);
  return u.v;
}

// LDS byte offset of a __shared__ object (generic addr low 32 bits == LDS addr)
__device__ __forceinline__ unsigned lds_off(const void* p) {
  return (unsigned)(unsigned long long)p;
}

// ---------------------------------------------------------------------------
// TDM: load a 128-row x 32-col bf16 tile (row stride ldk elements) from global
// into LDS with 16B padding per 64B row -> LDS row pitch 40 elements (80B),
// making 16-row b128 fragment reads bank-conflict-free.
// D# packed per CDNA5 ISA 8.3/8.4 (2D tensor, groups 2/3 zero).
// ---------------------------------------------------------------------------
__device__ __forceinline__ void tdm_load_tile_128x32(const bf16* gsrc,
                                                     unsigned lds_byte_off,
                                                     int ldk) {
  unsigned long long ga = (unsigned long long)gsrc;
  u32x4 g0;
  g0[0] = 1u;                                            // count=1 (valid D#)
  g0[1] = lds_byte_off;                                  // lds_addr
  g0[2] = (unsigned)ga;                                  // global_addr[31:0]
  g0[3] = (unsigned)((ga >> 32) & 0x1FFFFFFull)          // global_addr[56:32]
        | (2u << 30);                                    // type=2 ("image")
  unsigned td0 = (unsigned)ldk;                          // tensor_dim0 (elems)
  unsigned td1 = 128u;                                   // tensor_dim1 (rows)
  i32x8 g1;
  g1[0] = (int)((1u << 16)        // data_size = 2 bytes
              | (1u << 20)        // pad_enable
              | (3u << 22)        // pad_interval: 16 DWORDs (one 32-elem row)
              | (3u << 25));      // pad_amount: 4 DWORDs (16 bytes)
  g1[1] = (int)((td0 & 0xFFFFu) << 16);                  // tensor_dim0[15:0]
  g1[2] = (int)((td0 >> 16) | ((td1 & 0xFFFFu) << 16));  // dim0[31:16],dim1[15:0]
  g1[3] = (int)((td1 >> 16) | (32u << 16));              // dim1[31:16],tile_dim0=32
  g1[4] = (int)128u;                                     // tile_dim1=128,tile_dim2=0
  g1[5] = (int)td0;                                      // tensor_dim0_stride[31:0]
  g1[6] = 0;                                             // stride[47:32], dim1_stride lo
  g1[7] = 0;
  i32x4 z4; z4[0] = z4[1] = z4[2] = z4[3] = 0;
  i32x8 z8; for (int i = 0; i < 8; ++i) z8[i] = 0;
  __builtin_amdgcn_tensor_load_to_lds(g0, g1, z4, z4, z8, 0);
}

#define LDSP 40   // padded LDS row pitch in elements (80 bytes)

// ---------------------------------------------------------------------------
// fp32 -> bf16 conversion
// ---------------------------------------------------------------------------
__global__ __launch_bounds__(256)
void f32_to_bf16_kernel(const float* __restrict__ in, bf16* __restrict__ out,
                        unsigned long long n) {
  unsigned long long i = (unsigned long long)blockIdx.x * 256ull + threadIdx.x;
  if (i < n) out[i] = f2bf(in[i]);
}

// ---------------------------------------------------------------------------
// TDM-staged NT GEMM: C[M,N] = A[M,K]*B[N,K]^T, fp32 accum.
// Block 128x128, 256 threads = 8 waves (2x4), wave tile 64x32.
// Double-buffered LDS tiles filled by wave0 via tensor_load_to_lds.
// Requires M%128==0, N%128==0, K%32==0.
// ---------------------------------------------------------------------------
__global__ __launch_bounds__(256)
void gemm_nt_tdm(const bf16* __restrict__ A, const bf16* __restrict__ Bm,
                 int M, int N, int K,
                 float* __restrict__ Cf, bf16* __restrict__ Cb,
                 const float* __restrict__ bias) {
  __shared__ __align__(16) bf16 As[2][128 * LDSP];
  __shared__ __align__(16) bf16 Bs[2][128 * LDSP];
  const int lane = threadIdx.x & 31;
  const int w    = threadIdx.x >> 5;
  const int wm   = w >> 2;                 // 0..1 -> 64-row slice
  const int wn   = w & 3;                  // 0..3 -> 32-col slice
  const int row0 = blockIdx.y * 128;
  const int col0 = blockIdx.x * 128;

  v8f acc[4][2];
  for (int mt = 0; mt < 4; ++mt)
    for (int nt = 0; nt < 2; ++nt) acc[mt][nt] = v8f_zero();

  const int nk = K / 32;
  if (w == 0) {
    tdm_load_tile_128x32(A  + (size_t)row0 * K, lds_off(&As[0][0]), K);
    tdm_load_tile_128x32(Bm + (size_t)col0 * K, lds_off(&Bs[0][0]), K);
  }
  for (int kc = 0; kc < nk; ++kc) {
    const int buf = kc & 1;
    if (w == 0) {
      if (kc + 1 < nk) {
        tdm_load_tile_128x32(A  + (size_t)row0 * K + (kc + 1) * 32,
                             lds_off(&As[buf ^ 1][0]), K);
        tdm_load_tile_128x32(Bm + (size_t)col0 * K + (kc + 1) * 32,
                             lds_off(&Bs[buf ^ 1][0]), K);
        __builtin_amdgcn_s_wait_tensorcnt((short)2);  // stage kc landed (in-order)
      } else {
        __builtin_amdgcn_s_wait_tensorcnt((short)0);
      }
    }
    __syncthreads();                                   // release consumers
    const bf16* Ab = &As[buf][0];
    const bf16* Bb = &Bs[buf][0];
    v16bf af[4], bfr[2];
    for (int mt = 0; mt < 4; ++mt)
      af[mt] = load_frag_nt(Ab + (wm * 64 + mt * 16) * LDSP, LDSP, lane);
    for (int nt = 0; nt < 2; ++nt)
      bfr[nt] = load_frag_nt(Bb + (wn * 32 + nt * 16) * LDSP, LDSP, lane);
    for (int mt = 0; mt < 4; ++mt)
      for (int nt = 0; nt < 2; ++nt)
        acc[mt][nt] = wmma_bf16(af[mt], bfr[nt], acc[mt][nt]);
    __syncthreads();                                   // protect buf reuse
  }

  for (int mt = 0; mt < 4; ++mt)
    for (int nt = 0; nt < 2; ++nt) {
      int col   = col0 + wn * 32 + nt * 16 + (lane & 15);
      float bv  = bias ? bias[col] : 0.f;
      int rbase = row0 + wm * 64 + mt * 16 + (lane >> 4) * 8;
      for (int r = 0; r < 8; ++r) {
        float v = acc[mt][nt][r] + bv;
        size_t idx = (size_t)(rbase + r) * N + col;
        if (Cf) Cf[idx] = v;
        if (Cb) Cb[idx] = f2bf(v);
      }
    }
}

// ---------------------------------------------------------------------------
// Direct-load NT GEMM (for N%64 tiles, used by kv_down N=576).
// Block 128x64, 8 waves (4x2), wave tile 32x32.
// ---------------------------------------------------------------------------
__global__ __launch_bounds__(256)
void gemm_nt_bf16(const bf16* __restrict__ A, const bf16* __restrict__ Bm,
                  int M, int N, int K,
                  float* __restrict__ Cf, bf16* __restrict__ Cb,
                  const float* __restrict__ bias) {
  const int lane = threadIdx.x & 31;
  const int w    = threadIdx.x >> 5;
  const int wm   = w >> 1;
  const int wn   = w & 1;
  const int row0 = blockIdx.y * 128 + wm * 32;
  const int col0 = blockIdx.x * 64  + wn * 32;

  v8f acc[2][2];
  acc[0][0] = v8f_zero(); acc[0][1] = v8f_zero();
  acc[1][0] = v8f_zero(); acc[1][1] = v8f_zero();

  for (int k0 = 0; k0 < K; k0 += 32) {
    v16bf a0 = load_frag_nt(A  + (size_t)(row0 +  0) * K + k0, K, lane);
    v16bf a1 = load_frag_nt(A  + (size_t)(row0 + 16) * K + k0, K, lane);
    v16bf b0 = load_frag_nt(Bm + (size_t)(col0 +  0) * K + k0, K, lane);
    v16bf b1 = load_frag_nt(Bm + (size_t)(col0 + 16) * K + k0, K, lane);
    acc[0][0] = wmma_bf16(a0, b0, acc[0][0]);
    acc[0][1] = wmma_bf16(a0, b1, acc[0][1]);
    acc[1][0] = wmma_bf16(a1, b0, acc[1][0]);
    acc[1][1] = wmma_bf16(a1, b1, acc[1][1]);
  }

  for (int mt = 0; mt < 2; ++mt)
    for (int nt = 0; nt < 2; ++nt) {
      int col   = col0 + nt * 16 + (lane & 15);
      float bv  = bias ? bias[col] : 0.f;
      int rbase = row0 + mt * 16 + (lane >> 4) * 8;
      for (int r = 0; r < 8; ++r) {
        float v = acc[mt][nt][r] + bv;
        size_t idx = (size_t)(rbase + r) * N + col;
        if (Cf) Cf[idx] = v;
        if (Cb) Cb[idx] = f2bf(v);
      }
    }
}

// ---------------------------------------------------------------------------
// RMSNorm over a row slice: out_bf16 = x * rsqrt(mean(x^2)+eps) * w
// ---------------------------------------------------------------------------
__global__ __launch_bounds__(256)
void rmsnorm_kernel(const float* __restrict__ in, int instride, int width,
                    const float* __restrict__ w, bf16* __restrict__ out,
                    int outstride) {
  __shared__ float red[8];
  __shared__ float total;
  const int row = blockIdx.x;
  const float* x = in + (size_t)row * instride;
  float ss = 0.f;
  for (int i = threadIdx.x; i < width; i += 256) { float v = x[i]; ss += v * v; }
  for (int m = 16; m >= 1; m >>= 1) ss += __shfl_xor(ss, m, 32);
  if ((threadIdx.x & 31) == 0) red[threadIdx.x >> 5] = ss;
  __syncthreads();
  if (threadIdx.x == 0) {
    float t = 0.f; for (int i = 0; i < 8; ++i) t += red[i];
    total = t;
  }
  __syncthreads();
  const float scale = rsqrtf(total / (float)width + 1.1920929e-07f);
  bf16* o = out + (size_t)row * outstride;
  for (int i = threadIdx.x; i < width; i += 256) o[i] = f2bf(x[i] * scale * w[i]);
}

// ---------------------------------------------------------------------------
// RoPE for q: qup fp32 [M, H*QKD] -> q bf16 [B,H,S,QKD], 1/sqrt(QKD) folded in.
// ---------------------------------------------------------------------------
__global__ __launch_bounds__(192)
void rope_q_kernel(const float* __restrict__ qup, const float* __restrict__ fc,
                   const float* __restrict__ fs, bf16* __restrict__ qout) {
  const int bi = blockIdx.x;
  const int h = bi % Hn, row = bi / Hn;
  const int s = row % Ssz, b = row / Ssz;
  const int d = threadIdx.x;
  const float* qr = qup + (size_t)row * (Hn * QKD) + h * QKD;
  float v;
  if (d < NOPE) {
    v = qr[d];
  } else {
    int p = (d - NOPE) >> 1;
    float c = fc[s * 32 + p], sn = fs[s * 32 + p];
    float x0 = qr[NOPE + 2 * p], x1 = qr[NOPE + 2 * p + 1];
    v = ((d & 1) == 0) ? (x0 * c - x1 * sn) : (x0 * sn + x1 * c);
  }
  const float sc = 0.0721687836487032f;  // 1/sqrt(192)
  size_t bh = (size_t)(b * Hn + h);
  qout[(bh * Ssz + s) * QKD + d] = f2bf(v * sc);
}

// ---------------------------------------------------------------------------
// RoPE for shared k_rope: kvdown fp32 [M,576] (first 64 cols) -> bf16 [M,64]
// ---------------------------------------------------------------------------
__global__ __launch_bounds__(64)
void rope_k_kernel(const float* __restrict__ kvdown, const float* __restrict__ fc,
                   const float* __restrict__ fs, bf16* __restrict__ krope) {
  const int row = blockIdx.x;
  const int s = row % Ssz;
  const int d = threadIdx.x;
  const float* kr = kvdown + (size_t)row * (ROPEd + KRr);
  int p = d >> 1;
  float c = fc[s * 32 + p], sn = fs[s * 32 + p];
  float x0 = kr[2 * p], x1 = kr[2 * p + 1];
  float v = ((d & 1) == 0) ? (x0 * c - x1 * sn) : (x0 * sn + x1 * c);
  krope[(size_t)row * ROPEd + d] = f2bf(v);
}

// ---------------------------------------------------------------------------
// Assemble K bf16 [B,H,S,QKD] and Vt bf16 [B,H,VD,S].
// ---------------------------------------------------------------------------
__global__ __launch_bounds__(192)
void assemble_kv_kernel(const float* __restrict__ kvup,
                        const bf16* __restrict__ krope,
                        bf16* __restrict__ Kb, bf16* __restrict__ Vt) {
  const int bi = blockIdx.x;
  const int h = bi % Hn, row = bi / Hn;
  const int s = row % Ssz, b = row / Ssz;
  const int d = threadIdx.x;
  size_t bh = (size_t)(b * Hn + h);
  const float* kv = kvup + (size_t)row * (Hn * (NOPE + VDd)) + h * (NOPE + VDd);
  if (d < NOPE) {
    Kb[(bh * Ssz + s) * QKD + d] = f2bf(kv[d]);
    Vt[(bh * VDd + d) * Ssz + s] = f2bf(kv[NOPE + d]);
  } else {
    Kb[(bh * Ssz + s) * QKD + d] = krope[(size_t)row * ROPEd + (d - NOPE)];
  }
}

// ---------------------------------------------------------------------------
// Streaming causal attention (one wave per 16-query tile, 32 keys/step).
// ---------------------------------------------------------------------------
__global__ __launch_bounds__(256)
void mla_attention_kernel(const bf16* __restrict__ Q, const bf16* __restrict__ Kb,
                          const bf16* __restrict__ Vt, bf16* __restrict__ Ob) {
  __shared__ __align__(16) bf16 plds[8][16 * 32];
  const int lane = threadIdx.x & 31;
  const int w    = threadIdx.x >> 5;
  const int tiles_per_bh = Ssz / 16;                 // 128
  const int t  = blockIdx.x * 8 + w;
  const int bh = t / tiles_per_bh;
  const int qt = t % tiles_per_bh;

  const bf16* qbase = Q  + ((size_t)bh * Ssz + (size_t)qt * 16) * QKD;
  const bf16* kbase = Kb + (size_t)bh * Ssz * QKD;
  const bf16* vbase = Vt + (size_t)bh * VDd * Ssz;

  v16bf qf[6];
  for (int kk = 0; kk < 6; ++kk) qf[kk] = load_frag_nt(qbase + kk * 32, QKD, lane);

  v8f o[8];
  for (int nt = 0; nt < 8; ++nt) o[nt] = v8f_zero();
  float mrow[8], lrow[8];
  for (int r = 0; r < 8; ++r) { mrow[r] = -3.0e38f; lrow[r] = 0.f; }

  const int halfsel = lane >> 4;
  const int qrow0   = qt * 16 + halfsel * 8;
  const int nk = ((qt + 1) * 16 + 31) >> 5;

  bf16* pl = &plds[w][0];

  for (int kc = 0; kc < nk; ++kc) {
    const int kb = kc * 32;
    if (kc + 1 < nk)   // prefetch next key chunk (global_prefetch_b8)
      __builtin_prefetch(kbase + (size_t)(kb + 32) * QKD + lane * QKD, 0, 0);
    v8f s0 = v8f_zero(), s1 = v8f_zero();
    for (int kk = 0; kk < 6; ++kk) {
      v16bf k0 = load_frag_nt(kbase + (size_t)(kb +  0) * QKD + kk * 32, QKD, lane);
      v16bf k1 = load_frag_nt(kbase + (size_t)(kb + 16) * QKD + kk * 32, QKD, lane);
      s0 = wmma_bf16(qf[kk], k0, s0);
      s1 = wmma_bf16(qf[kk], k1, s1);
    }
    const int key0 = kb + (lane & 15);
    for (int r = 0; r < 8; ++r) {
      int qi = qrow0 + r;
      if (key0      > qi) s0[r] = -3.0e38f;
      if (key0 + 16 > qi) s1[r] = -3.0e38f;
    }
    float alpha[8];
    for (int r = 0; r < 8; ++r) {
      float mx = fmaxf(s0[r], s1[r]);
      for (int m = 8; m >= 1; m >>= 1) mx = fmaxf(mx, __shfl_xor(mx, m, 32));
      float mnew = fmaxf(mrow[r], mx);
      alpha[r] = __expf(mrow[r] - mnew);
      float p0 = __expf(s0[r] - mnew);
      float p1 = __expf(s1[r] - mnew);
      float ps = p0 + p1;
      for (int m = 8; m >= 1; m >>= 1) ps += __shfl_xor(ps, m, 32);
      lrow[r] = lrow[r] * alpha[r] + ps;
      mrow[r] = mnew;
      s0[r] = p0; s1[r] = p1;
    }
    for (int nt = 0; nt < 8; ++nt)
      for (int r = 0; r < 8; ++r) o[nt][r] *= alpha[r];

    for (int r = 0; r < 8; ++r) {
      int rr = halfsel * 8 + r;
      pl[rr * 32 +      (lane & 15)] = f2bf(s0[r]);
      pl[rr * 32 + 16 + (lane & 15)] = f2bf(s1[r]);
    }
    asm volatile("" ::: "memory");
    v16bf pf;
    {
      int row = lane & 15;
      int ko  = (lane >> 4) * 8;
      union { v16bf v; v8bf h[2]; } u;
      u.h[0] = *(const v8bf*)(pl + row * 32 + ko);
      u.h[1] = *(const v8bf*)(pl + row * 32 + ko + 16);
      pf = u.v;
    }
    asm volatile("" ::: "memory");
    for (int nt = 0; nt < 8; ++nt) {
      v16bf vf = load_frag_nt(vbase + (size_t)(nt * 16) * Ssz + kb, Ssz, lane);
      o[nt] = wmma_bf16(pf, vf, o[nt]);
    }
  }

  const int b = bh / Hn, h = bh % Hn;
  for (int r = 0; r < 8; ++r) {
    float inv = 1.f / lrow[r];
    int s = qt * 16 + halfsel * 8 + r;
    bf16* orow = Ob + ((size_t)(b * Ssz + s)) * (Hn * VDd) + h * VDd;
    for (int nt = 0; nt < 8; ++nt)
      orow[nt * 16 + (lane & 15)] = f2bf(o[nt][r] * inv);
  }
}

// ---------------------------------------------------------------------------
// Host orchestration
// ---------------------------------------------------------------------------
extern "C" void kernel_launch(void* const* d_in, const int* in_sizes, int n_in,
                              void* d_out, int out_size, void* d_ws, size_t ws_size,
                              hipStream_t stream) {
  (void)in_sizes; (void)n_in; (void)out_size; (void)ws_size;
  const float* x        = (const float*)d_in[0];
  const float* fc       = (const float*)d_in[1];
  const float* fs       = (const float*)d_in[2];
  const float* Wq_down  = (const float*)d_in[3];
  const float* q_norm_w = (const float*)d_in[4];
  const float* Wq_up    = (const float*)d_in[5];
  const float* Wkv_down = (const float*)d_in[6];
  const float* kv_norm_w= (const float*)d_in[7];
  const float* Wkv_up   = (const float*)d_in[8];
  const float* Wo       = (const float*)d_in[9];
  const float* bo       = (const float*)d_in[10];
  float* out = (float*)d_out;

  char* wsp = (char*)d_ws;
  auto alloc = [&](size_t bytes) -> void* {
    void* p = (void*)wsp;
    wsp += (bytes + 255) & ~(size_t)255;
    return p;
  };
  bf16* xb      = (bf16*)alloc((size_t)Mrows * Dm * 2);
  bf16* wqd     = (bf16*)alloc((size_t)QRr * Dm * 2);
  bf16* wqu     = (bf16*)alloc((size_t)(Hn * QKD) * QRr * 2);
  bf16* wkvd    = (bf16*)alloc((size_t)(ROPEd + KRr) * Dm * 2);
  bf16* wkvu    = (bf16*)alloc((size_t)(Hn * (NOPE + VDd)) * KRr * 2);
  bf16* wob     = (bf16*)alloc((size_t)Dm * (Hn * VDd) * 2);
  bf16* qnorm_b = (bf16*)alloc((size_t)Mrows * QRr * 2);
  bf16* q_b     = (bf16*)alloc((size_t)Bsz * Hn * Ssz * QKD * 2);
  bf16* krope_b = (bf16*)alloc((size_t)Mrows * ROPEd * 2);
  bf16* ckv_b   = (bf16*)alloc((size_t)Mrows * KRr * 2);
  bf16* k_b     = (bf16*)alloc((size_t)Bsz * Hn * Ssz * QKD * 2);
  bf16* vt_b    = (bf16*)alloc((size_t)Bsz * Hn * VDd * Ssz * 2);
  bf16* o_b     = (bf16*)alloc((size_t)Mrows * (Hn * VDd) * 2);
  float* fpool  = (float*)alloc((size_t)Mrows * (Hn * (NOPE + VDd)) * 4);

  auto conv = [&](const float* src, bf16* dst, size_t n) {
    f32_to_bf16_kernel<<<dim3((unsigned)((n + 255) / 256)), dim3(256), 0, stream>>>(
        src, dst, (unsigned long long)n);
  };
  conv(x,        xb,   (size_t)Mrows * Dm);
  conv(Wq_down,  wqd,  (size_t)QRr * Dm);
  conv(Wq_up,    wqu,  (size_t)(Hn * QKD) * QRr);
  conv(Wkv_down, wkvd, (size_t)(ROPEd + KRr) * Dm);
  conv(Wkv_up,   wkvu, (size_t)(Hn * (NOPE + VDd)) * KRr);
  conv(Wo,       wob,  (size_t)Dm * (Hn * VDd));

  dim3 blk(256);

  // ---- q path ----
  float* qdown = fpool;
  gemm_nt_tdm<<<dim3(QRr / 128, Mrows / 128), blk, 0, stream>>>(
      xb, wqd, Mrows, QRr, Dm, qdown, nullptr, nullptr);
  rmsnorm_kernel<<<dim3(Mrows), blk, 0, stream>>>(
      qdown, QRr, QRr, q_norm_w, qnorm_b, QRr);
  float* qup = fpool;  // qdown dead
  gemm_nt_tdm<<<dim3((Hn * QKD) / 128, Mrows / 128), blk, 0, stream>>>(
      qnorm_b, wqu, Mrows, Hn * QKD, QRr, qup, nullptr, nullptr);
  rope_q_kernel<<<dim3(Mrows * Hn), dim3(192), 0, stream>>>(qup, fc, fs, q_b);

  // ---- kv path ----
  float* kvdown = fpool;  // qup dead
  gemm_nt_bf16<<<dim3((ROPEd + KRr) / 64, Mrows / 128), blk, 0, stream>>>(
      xb, wkvd, Mrows, ROPEd + KRr, Dm, kvdown, nullptr, nullptr);
  rope_k_kernel<<<dim3(Mrows), dim3(64), 0, stream>>>(kvdown, fc, fs, krope_b);
  rmsnorm_kernel<<<dim3(Mrows), blk, 0, stream>>>(
      kvdown + ROPEd, ROPEd + KRr, KRr, kv_norm_w, ckv_b, KRr);
  float* kvup = fpool;  // kvdown dead
  gemm_nt_tdm<<<dim3((Hn * (NOPE + VDd)) / 128, Mrows / 128), blk, 0, stream>>>(
      ckv_b, wkvu, Mrows, Hn * (NOPE + VDd), KRr, kvup, nullptr, nullptr);
  assemble_kv_kernel<<<dim3(Mrows * Hn), dim3(192), 0, stream>>>(
      kvup, krope_b, k_b, vt_b);

  // ---- attention ----
  const int qtiles = Bsz * Hn * (Ssz / 16);
  mla_attention_kernel<<<dim3(qtiles / 8), blk, 0, stream>>>(q_b, k_b, vt_b, o_b);

  // ---- output projection (+bias) ----
  gemm_nt_tdm<<<dim3(Dm / 128, Mrows / 128), blk, 0, stream>>>(
      o_b, wob, Mrows, Dm, Hn * VDd, out, nullptr, bo);
}
